// DenseGATLayer_54443005444823
// MI455X (gfx1250) — matrix operand order repaired and checked by reading, not compiled, not run on hardware
//
#include <hip/hip_runtime.h>
#include <hip/hip_bf16.h>

typedef __attribute__((ext_vector_type(2)))  float  v2f;
typedef __attribute__((ext_vector_type(8)))  float  v8f;
typedef __attribute__((ext_vector_type(16))) __bf16 v16bf;

#define NB   4
#define NN   2048
#define FINC 128
#define KH   4
#define DH   32
#define KD   128   // KH*DH
#define NEG_SLOPE 0.2f

// ---------------------------------------------------------------------------
// Kernel 1: Wh = H @ W  (full fp32 via v_wmma_f32_16x16x4_f32)
// Grid: 512 blocks x 256 threads. Wave w of a block handles n-tile w of the
// block's m-tile. Writes Wh (f32, staged in d_out) and WhT (bf16, [B,K,D,N]).
// ---------------------------------------------------------------------------
__global__ __launch_bounds__(256) void gat_wh_gemm(const float* __restrict__ H,
                                                   const float* __restrict__ W,
                                                   float* __restrict__ Wh,
                                                   __bf16* __restrict__ WhT) {
  const int lane = threadIdx.x & 31;
  const int nt   = threadIdx.x >> 5;   // n-tile 0..7 (16 cols each -> 128)
  const int mt   = blockIdx.x;         // m-tile 0..511 (16 rows each -> 8192)
  const int m    = lane & 15;
  const int grp  = lane >> 4;

  const float* hrow = H + (size_t)(mt * 16 + m) * FINC;  // A: row m per lane
  const float* wcol = W + nt * 16 + m;                   // B: col n per lane

  v8f acc = {};
  for (int k = 0; k < FINC; k += 4) {
    // A 16x4 f32: lane(m,grp): elems = A[m][2*grp + {0,1}] at k-base
    v2f a = *(const v2f*)(hrow + k + 2 * grp);
    // B 4x16 f32: lane(m,grp): elems = B[2*grp + {0,1}][n]
    v2f b;
    b.x = wcol[(size_t)(k + 2 * grp) * KD];
    b.y = wcol[(size_t)(k + 2 * grp + 1) * KD];
    acc = __builtin_amdgcn_wmma_f32_16x16x4_f32(false, a, false, b,
                                                (short)0, acc, false, false);
  }

  const int kd = nt * 16 + m;          // C col n -> output feature index
  const int kk = kd >> 5;              // head
  const int dd = kd & 31;              // dim within head
#pragma unroll
  for (int r = 0; r < 8; ++r) {
    const int node = mt * 16 + r + 8 * grp;   // C row = r + 8*grp
    Wh[(size_t)node * KD + kd] = acc[r];
    const int b_ = node >> 11;                // node / 2048
    const int n_ = node & (NN - 1);
    WhT[((size_t)((b_ * KH + kk) * DH + dd)) * NN + n_] = (__bf16)acc[r];
  }
}

// ---------------------------------------------------------------------------
// Kernel 2: s[b,k,n] = Wh[b,n,k,:]·a_src[k,:]  (and t with a_dst), stored
// transposed [B,K,N] so the attention kernel loads them contiguously.
// ---------------------------------------------------------------------------
__global__ __launch_bounds__(256) void gat_scores(const float* __restrict__ Wh,
                                                  const float* __restrict__ a_src,
                                                  const float* __restrict__ a_dst,
                                                  float* __restrict__ s_t,
                                                  float* __restrict__ t_t) {
  const int idx = blockIdx.x * 256 + threadIdx.x;  // B*N*K = 32768
  const int k = idx & (KH - 1);
  const int n = (idx >> 2) & (NN - 1);
  const int b = idx >> 13;
  const float* w = Wh + ((size_t)(b * NN + n)) * KD + k * DH;
  float s = 0.f, t = 0.f;
#pragma unroll
  for (int d = 0; d < DH; ++d) {
    const float v = w[d];
    s += v * a_src[k * DH + d];
    t += v * a_dst[k * DH + d];
  }
  s_t[(size_t)(b * KH + k) * NN + n] = s;
  t_t[(size_t)(b * KH + k) * NN + n] = t;
}

// ---------------------------------------------------------------------------
// Kernel 3: fused masked softmax + aggregation (flash-style, no running max —
// scores are O(1) so exp() is safe). One wave per (b, head, 16-row i-tile).
// P tile built directly in the bf16 A-operand layout; two
// v_wmma_f32_16x16x32_bf16 per 32-wide j-tile (d-tiles 0/1).
// ---------------------------------------------------------------------------
__global__ __launch_bounds__(256) void gat_attn(const float* __restrict__ A,
                                                const __bf16* __restrict__ WhT,
                                                const float* __restrict__ s_t,
                                                const float* __restrict__ t_t,
                                                float* __restrict__ out) {
  const int lane = threadIdx.x & 31;
  const int wid  = blockIdx.x * 8 + (threadIdx.x >> 5);  // 0..2047
  const int it   = wid & 127;          // i-tile
  const int k    = (wid >> 7) & (KH - 1);
  const int b    = wid >> 9;
  const int m    = lane & 15;
  const int grp  = lane >> 4;
  const int i0   = it * 16;

  const float  si   = s_t[(size_t)(b * KH + k) * NN + i0 + m];
  const float* trow = t_t + (size_t)(b * KH + k) * NN;
  const float* arow = A + ((size_t)b * NN + (i0 + m)) * NN;
  const __bf16* w0  = WhT + (size_t)((b * KH + k) * DH + m) * NN;        // d = m
  const __bf16* w1  = WhT + (size_t)((b * KH + k) * DH + 16 + m) * NN;   // d = 16+m

  v8f acc0 = {}, acc1 = {};
  float psum = 0.f;

  for (int jt = 0; jt < NN; jt += 32) {
    // A-operand K map (16-bit A 16x32): h<8 -> K=8*grp+h ; h>=8 -> K=16+8*grp+(h-8)
    const int c0 = jt + 8 * grp;
    const int c1 = jt + 16 + 8 * grp;
    float av[16], tv[16];
    *(float4*)&av[0]  = *(const float4*)(arow + c0);
    *(float4*)&av[4]  = *(const float4*)(arow + c0 + 4);
    *(float4*)&av[8]  = *(const float4*)(arow + c1);
    *(float4*)&av[12] = *(const float4*)(arow + c1 + 4);
    *(float4*)&tv[0]  = *(const float4*)(trow + c0);
    *(float4*)&tv[4]  = *(const float4*)(trow + c0 + 4);
    *(float4*)&tv[8]  = *(const float4*)(trow + c1);
    *(float4*)&tv[12] = *(const float4*)(trow + c1 + 4);

    v16bf p;
#pragma unroll
    for (int h = 0; h < 16; ++h) {
      float x = si + tv[h];
      x = x > 0.f ? x : NEG_SLOPE * x;                // leaky relu
      const float pv = (av[h] > 0.f) ? __expf(x) : 0.f;  // mask: A>0
      psum += pv;
      p[h] = (__bf16)pv;
    }

    // B operand (32x16 bf16): elem h of lane(m,grp) = Wh[j = c0+h (h<8) / c1+h-8 (h>=8)][d]
    typedef __attribute__((ext_vector_type(8))) __bf16 v8bf_;
    const v8bf_ b0l = *(const v8bf_*)(w0 + c0), b0h = *(const v8bf_*)(w0 + c1);
    const v8bf_ b1l = *(const v8bf_*)(w1 + c0), b1h = *(const v8bf_*)(w1 + c1);
    const v16bf b0 = __builtin_shufflevector(b0l, b0h, 0,1,2,3,4,5,6,7,8,9,10,11,12,13,14,15);
    const v16bf b1 = __builtin_shufflevector(b1l, b1h, 0,1,2,3,4,5,6,7,8,9,10,11,12,13,14,15);
    acc0 = __builtin_amdgcn_wmma_f32_16x16x32_bf16(false, p, false, b0,
                                                   (short)0, acc0, false, false);
    acc1 = __builtin_amdgcn_wmma_f32_16x16x32_bf16(false, p, false, b1,
                                                   (short)0, acc1, false, false);
  }

  // Combine the two half-lane partial row sums: lane l covered half the j's of
  // row m; lane l^16 covered the other half.
  psum += __shfl_xor(psum, 16, 32);

#pragma unroll
  for (int r = 0; r < 8; ++r) {
    const int row = r + 8 * grp;                       // C row for this vgpr
    const float rs = __shfl(psum, row, 32);            // full row sum of row
    const float inv = (rs > 0.f) ? 1.f / rs : 0.f;
    float* orow = out + (size_t)(b * NN + i0 + row) * KD + k * DH;
    orow[m]      = acc0[r] * inv;
    orow[16 + m] = acc1[r] * inv;
  }
}

// ---------------------------------------------------------------------------
// Launch. d_out (4 MB) doubles as the f32 Wh staging buffer (kernel 3 fully
// overwrites it). Workspace: WhT bf16 (2 MB) + s_t/t_t (128 KB each).
// ---------------------------------------------------------------------------
extern "C" void kernel_launch(void* const* d_in, const int* in_sizes, int n_in,
                              void* d_out, int out_size, void* d_ws, size_t ws_size,
                              hipStream_t stream) {
  const float* H     = (const float*)d_in[0];
  const float* A     = (const float*)d_in[1];
  const float* W     = (const float*)d_in[2];
  const float* a_src = (const float*)d_in[3];
  const float* a_dst = (const float*)d_in[4];
  float* out = (float*)d_out;

  char* ws = (char*)d_ws;
  __bf16* WhT = (__bf16*)ws;                                   // 2 MB
  float*  s_t = (float*)(ws + 2u * 1024u * 1024u);             // 128 KB
  float*  t_t = (float*)(ws + 2u * 1024u * 1024u + 131072u);   // 128 KB
  float*  Wh  = out;  // f32 staging, fully overwritten by gat_attn

  gat_wh_gemm<<<512, 256, 0, stream>>>(H, W, Wh, WhT);
  gat_scores<<<(NB * NN * KH) / 256, 256, 0, stream>>>(Wh, a_src, a_dst, s_t, t_t);
  gat_attn<<<256, 256, 0, stream>>>(A, WhT, s_t, t_t, out);
}
